// GAT_69630009802899
// MI455X (gfx1250) — matrix operand order repaired and checked by reading, not compiled, not run on hardware
//
#include <hip/hip_runtime.h>

typedef _Float16 f16;
typedef __attribute__((ext_vector_type(16))) _Float16 v16h;
typedef __attribute__((ext_vector_type(8)))  float    v8f;

#define NEG_SLOPE 0.2f

__device__ __forceinline__ unsigned int fkey(float f) {
  unsigned int b = __float_as_uint(f);
  return (b & 0x80000000u) ? ~b : (b | 0x80000000u);
}
__device__ __forceinline__ float funkey(unsigned int k) {
  return __uint_as_float((k & 0x80000000u) ? (k & 0x7FFFFFFFu) : ~k);
}
__device__ __forceinline__ float lrelu(float v) { return v > 0.f ? v : NEG_SLOPE * v; }

__device__ __forceinline__ void edge_sd(const int* __restrict__ ei, int e, int E, int& s, int& d) {
  if (e < E) { s = ei[e]; d = ei[E + e]; }
  else       { s = e - E; d = s; }           // self-loops appended
}

// ---------------- init ----------------
__global__ void fill_zero_u32(unsigned int* p, long long n) {
  long long i = (long long)blockIdx.x * blockDim.x + threadIdx.x;
  if (i < n) p[i] = 0u;
}

// ---------------- pack weights into WMMA B-fragment layout ----------------
// B 32x16 f16 order used here (as A): lanes 0-15 hold K=0..15 of column N=lane; lanes 16-31 hold
// K=16..31 of column N=lane-16. Each fragment = 32 lanes x 16 halfs, contiguous.
__global__ void pack_w(const float* __restrict__ W1, const float* __restrict__ W2,
                       f16* __restrict__ w1f, f16* __restrict__ w2f) {
  int idx = blockIdx.x * blockDim.x + threadIdx.x;
  if (idx < 1024) {                       // W1: 8 ksteps * 4 ntiles * 32 lanes
    int lane = idx & 31;
    int t    = idx >> 5;
    int nt   = t & 3;
    int kk   = t >> 2;
    int n    = nt * 16 + (lane & 15);
    int kb   = kk * 32 + ((lane >> 4) * 16);
    f16* o = w1f + (long long)idx * 16;
#pragma unroll
    for (int i = 0; i < 16; ++i) o[i] = (f16)W1[(kb + i) * 64 + n];
  } else if (idx < 1024 + 192) {          // W2: 2 ksteps * 3 ntiles(pad 48) * 32 lanes
    int j    = idx - 1024;
    int lane = j & 31;
    int t    = j >> 5;
    int nt   = t % 3;
    int kk   = t / 3;
    int n    = nt * 16 + (lane & 15);
    int kb   = kk * 32 + ((lane >> 4) * 16);
    f16* o = w2f + (long long)j * 16;
#pragma unroll
    for (int i = 0; i < 16; ++i) o[i] = (n < 40) ? (f16)W2[(kb + i) * 40 + n] : (f16)0.f;
  }
}

// ---------------- WMMA GEMM: out[nrows x ncols] = A[nrows x (KK*32)] * W ----------------
// One wave computes a 16-row stripe. A fragments converted f32->f16 on the fly.
// B fragments are staged once per block into LDS with gfx1250 async copies
// (GLOBAL_LOAD_ASYNC_TO_LDS_B128, ASYNCcnt), then read back as ds_load_b128.
// A 16x32 f16 order used here (as B): lane<16 -> M=lane, K=0..15; lane>=16 -> M=lane-16, K=16..31.
template <int KK, int NT>
__global__ void gemm_wmma(const float* __restrict__ A, const f16* __restrict__ wfrag,
                          float* __restrict__ out, int nrows, int lda, int ldo, int ncols) {
  constexpr int NFRAG = KK * NT;
  __shared__ alignas(32) f16 sfrag[NFRAG * 32 * 16];   // one fragment = 32 lanes x 16 halfs

  // ---- cooperative async copy of all B fragments into LDS (16B per transfer) ----
  {
    constexpr int NXFER = NFRAG * 32 * 16 * 2 / 16;    // b128 transfers
    for (int i = threadIdx.x; i < NXFER; i += blockDim.x) {
      unsigned           lds = (unsigned)(unsigned long long)(&sfrag[0]) + i * 16;
      unsigned long long g   = (unsigned long long)(wfrag) + (unsigned long long)i * 16;
#if defined(__gfx1250__)
      asm volatile("global_load_async_to_lds_b128 %0, %1, off"
                   :: "v"(lds), "v"(g) : "memory");
#else
      *(float4*)(&sfrag[0] + i * 8) = *(const float4*)((const char*)wfrag + i * 16);
#endif
    }
#if defined(__gfx1250__)
    asm volatile("s_wait_asynccnt 0x0" ::: "memory");
#endif
    __syncthreads();
  }

  int wave  = threadIdx.x >> 5;
  int lane  = threadIdx.x & 31;
  int rtile = blockIdx.x * (blockDim.x >> 5) + wave;
  int ntr   = (nrows + 15) >> 4;
  if (rtile >= ntr) return;               // wave-uniform: EXEC stays all-ones

  int row = rtile * 16 + (lane & 15);
  if (row >= nrows) row = nrows - 1;      // clamp instead of predicating (WMMA needs full EXEC)
  int kbase = (lane >> 4) * 16;

  v8f acc[NT] = {};
  const float* arow = A + (long long)row * lda;

#pragma unroll
  for (int kk = 0; kk < KK; ++kk) {
    const float* p = arow + kk * 32 + kbase;
    float4 f0 = *(const float4*)(p);
    float4 f1 = *(const float4*)(p + 4);
    float4 f2 = *(const float4*)(p + 8);
    float4 f3 = *(const float4*)(p + 12);
    v16h a;
    a[0]  = (f16)f0.x; a[1]  = (f16)f0.y; a[2]  = (f16)f0.z; a[3]  = (f16)f0.w;
    a[4]  = (f16)f1.x; a[5]  = (f16)f1.y; a[6]  = (f16)f1.z; a[7]  = (f16)f1.w;
    a[8]  = (f16)f2.x; a[9]  = (f16)f2.y; a[10] = (f16)f2.z; a[11] = (f16)f2.w;
    a[12] = (f16)f3.x; a[13] = (f16)f3.y; a[14] = (f16)f3.z; a[15] = (f16)f3.w;
#pragma unroll
    for (int t = 0; t < NT; ++t) {
      v16h b = *(const v16h*)(&sfrag[((kk * NT + t) * 32 + lane) * 16]);
      acc[t] = __builtin_amdgcn_wmma_f32_16x16x32_f16(false, a, false, b,
                                                      (short)0, acc[t], false, false);
    }
  }

  // C/D layout: VGPR j -> row j (lanes 0-15) / row j+8 (lanes 16-31); col = lane&15
  int rbase = rtile * 16 + ((lane >> 4) * 8);
  int c0    = lane & 15;
#pragma unroll
  for (int t = 0; t < NT; ++t) {
    int c = t * 16 + c0;
    if (c < ncols) {
#pragma unroll
      for (int j = 0; j < 8; ++j) {
        int r = rbase + j;
        if (r < nrows) out[(long long)r * ldo + c] = acc[t][j];
      }
    }
  }
}

// ---------------- attention dot products ----------------
__global__ void att1_k(const float* __restrict__ h1, const float* __restrict__ aw,
                       const float* __restrict__ dw, float* __restrict__ as,
                       float* __restrict__ ad, int N) {
  int i = blockIdx.x * blockDim.x + threadIdx.x;   // n*8+h
  if (i >= N * 8) return;
  int h = i & 7;
  const float* hp = h1 + (long long)(i >> 3) * 64 + h * 8;
  const float* sp = aw + h * 8;
  const float* dp = dw + h * 8;
  float s = 0.f, d = 0.f;
#pragma unroll
  for (int c = 0; c < 8; ++c) { float v = hp[c]; s += v * sp[c]; d += v * dp[c]; }
  as[i] = s; ad[i] = d;
}

__global__ void att2_k(const float* __restrict__ h2, const float* __restrict__ aw,
                       const float* __restrict__ dw, float* __restrict__ as,
                       float* __restrict__ ad, int N) {
  int n = blockIdx.x * blockDim.x + threadIdx.x;
  if (n >= N) return;
  const float* p = h2 + (long long)n * 40;
  float s = 0.f, d = 0.f;
#pragma unroll
  for (int c = 0; c < 40; ++c) { float v = p[c]; s += v * aw[c]; d += v * dw[c]; }
  as[n] = s; ad[n] = d;
}

// ---------------- edge phase: segment max / denom / aggregate ----------------
__global__ void emax_k(const int* __restrict__ ei, int E, int Etot,
                       const float* __restrict__ as, const float* __restrict__ ad,
                       unsigned int* __restrict__ mkey, int H) {
  long long i = (long long)blockIdx.x * blockDim.x + threadIdx.x;
  if (i >= (long long)Etot * H) return;
  int e = (int)(i / H), h = (int)(i % H);
  int s, d; edge_sd(ei, e, E, s, d);
  float lg = lrelu(as[(long long)s * H + h] + ad[(long long)d * H + h]);
  atomicMax(mkey + (long long)d * H + h, fkey(lg));
}

__global__ void eden_k(const int* __restrict__ ei, int E, int Etot,
                       const float* __restrict__ as, const float* __restrict__ ad,
                       const unsigned int* __restrict__ mkey, float* __restrict__ den, int H) {
  long long i = (long long)blockIdx.x * blockDim.x + threadIdx.x;
  if (i >= (long long)Etot * H) return;
  int e = (int)(i / H), h = (int)(i % H);
  int s, d; edge_sd(ei, e, E, s, d);
  float lg = lrelu(as[(long long)s * H + h] + ad[(long long)d * H + h]);
  long long di = (long long)d * H + h;
  atomicAdd(den + di, expf(lg - funkey(mkey[di])));
}

__global__ void eagg1_k(const int* __restrict__ ei, int E, int Etot,
                        const float* __restrict__ as, const float* __restrict__ ad,
                        const unsigned int* __restrict__ mkey, const float* __restrict__ den,
                        const float* __restrict__ h1, float* __restrict__ agg) {
  long long i = (long long)blockIdx.x * blockDim.x + threadIdx.x;
  if (i >= (long long)Etot * 8) return;
  int e = (int)(i >> 3), h = (int)(i & 7);
  int s, d; edge_sd(ei, e, E, s, d);
  float lg = lrelu(as[(long long)s * 8 + h] + ad[(long long)d * 8 + h]);
  long long di = (long long)d * 8 + h;
  float alpha = expf(lg - funkey(mkey[di])) / den[di];
  const float* hp = h1 + (long long)s * 64 + h * 8;
  float* op = agg + (long long)d * 64 + h * 8;
#pragma unroll
  for (int c = 0; c < 8; ++c) atomicAdd(op + c, hp[c] * alpha);
}

__global__ void eagg2_k(const int* __restrict__ ei, int E, int Etot,
                        const float* __restrict__ as, const float* __restrict__ ad,
                        const unsigned int* __restrict__ mkey, const float* __restrict__ den,
                        const float* __restrict__ h2, float* __restrict__ out2) {
  long long i = (long long)blockIdx.x * blockDim.x + threadIdx.x;
  if (i >= (long long)Etot * 5) return;
  int e = (int)(i / 5), g = (int)(i % 5);
  int s, d; edge_sd(ei, e, E, s, d);
  float lg = lrelu(as[s] + ad[d]);
  float alpha = expf(lg - funkey(mkey[d])) / den[d];
  const float* hp = h2 + (long long)s * 40 + g * 8;
  float* op = out2 + (long long)d * 40 + g * 8;
#pragma unroll
  for (int c = 0; c < 8; ++c) atomicAdd(op + c, hp[c] * alpha);
}

// ---------------- elementwise ----------------
__global__ void bias_relu_k(float* __restrict__ agg, const float* __restrict__ b1, long long n) {
  long long i = (long long)blockIdx.x * blockDim.x + threadIdx.x;
  if (i >= n) return;
  agg[i] = fmaxf(agg[i] + b1[i & 63], 0.f);
}

__global__ void final_ls_k(const float* __restrict__ out2, const float* __restrict__ b2,
                           float* __restrict__ y, int N) {
  int n = blockIdx.x * blockDim.x + threadIdx.x;
  if (n >= N) return;
  float v[40];
  float mx = -3.0e38f;
  const float* p = out2 + (long long)n * 40;
#pragma unroll
  for (int c = 0; c < 40; ++c) { v[c] = p[c] + b2[c]; mx = fmaxf(mx, v[c]); }
  float sum = 0.f;
#pragma unroll
  for (int c = 0; c < 40; ++c) sum += expf(v[c] - mx);
  float ls = logf(sum) + mx;
  float* o = y + (long long)n * 40;
#pragma unroll
  for (int c = 0; c < 40; ++c) o[c] = v[c] - ls;
}

// ---------------- launcher ----------------
static inline int nblk(long long n, int bs) { return (int)((n + bs - 1) / bs); }

extern "C" void kernel_launch(void* const* d_in, const int* in_sizes, int n_in,
                              void* d_out, int out_size, void* d_ws, size_t ws_size,
                              hipStream_t stream) {
  const float* x    = (const float*)d_in[0];
  const int*   ei   = (const int*)d_in[1];
  const float* W1   = (const float*)d_in[2];
  const float* as1w = (const float*)d_in[3];
  const float* ad1w = (const float*)d_in[4];
  const float* b1   = (const float*)d_in[5];
  const float* W2   = (const float*)d_in[6];
  const float* as2w = (const float*)d_in[7];
  const float* ad2w = (const float*)d_in[8];
  const float* b2   = (const float*)d_in[9];

  const int N    = in_sizes[0] / 256;
  const int E    = in_sizes[1] / 2;
  const int Etot = E + N;

  char* w = (char*)d_ws;
  auto alloc = [&](size_t bytes) -> char* {
    char* p = w;
    w += (bytes + 255) & ~(size_t)255;
    return p;
  };

  // non-zeroed scratch
  float* h1    = (float*)alloc((size_t)N * 64 * 4);
  float* as1   = (float*)alloc((size_t)N * 8 * 4);
  float* ad1   = (float*)alloc((size_t)N * 8 * 4);
  float* h2    = (float*)alloc((size_t)N * 40 * 4);
  float* as2   = (float*)alloc((size_t)N * 4);
  float* ad2   = (float*)alloc((size_t)N * 4);
  f16*   w1f   = (f16*)alloc((size_t)8 * 4 * 32 * 16 * 2);
  f16*   w2f   = (f16*)alloc((size_t)2 * 3 * 32 * 16 * 2);
  // zeroed scratch (contiguous region)
  char* zstart = w;
  unsigned int* m1   = (unsigned int*)alloc((size_t)N * 8 * 4);
  float*        den1 = (float*)alloc((size_t)N * 8 * 4);
  float*        agg1 = (float*)alloc((size_t)N * 64 * 4);   // becomes relu(h+bias) in place
  unsigned int* m2   = (unsigned int*)alloc((size_t)N * 4);
  float*        den2 = (float*)alloc((size_t)N * 4);
  float*        out2 = (float*)alloc((size_t)N * 40 * 4);
  long long zwords = (long long)(w - zstart) / 4;

  const int BS = 256;

  // 0) zero the accumulator/softmax-state region
  fill_zero_u32<<<nblk(zwords, BS), BS, 0, stream>>>((unsigned int*)zstart, zwords);

  // 1) pack W1/W2 into WMMA fragment layout
  pack_w<<<nblk(1024 + 192, BS), BS, 0, stream>>>(W1, W2, w1f, w2f);

  // 2) layer-1 GEMM: h1 = x @ W1   [N,256]x[256,64], K=8x32, N=4x16
  {
    int rtiles = (N + 15) / 16;
    gemm_wmma<8, 4><<<nblk(rtiles, 8), BS, 0, stream>>>(x, w1f, h1, N, 256, 64, 64);
  }

  // 3) attention dots, layer 1 (8 heads x 8 ch)
  att1_k<<<nblk((long long)N * 8, BS), BS, 0, stream>>>(h1, as1w, ad1w, as1, ad1, N);

  // 4) edge softmax + aggregate, layer 1
  emax_k<<<nblk((long long)Etot * 8, BS), BS, 0, stream>>>(ei, E, Etot, as1, ad1, m1, 8);
  eden_k<<<nblk((long long)Etot * 8, BS), BS, 0, stream>>>(ei, E, Etot, as1, ad1, m1, den1, 8);
  eagg1_k<<<nblk((long long)Etot * 8, BS), BS, 0, stream>>>(ei, E, Etot, as1, ad1, m1, den1, h1, agg1);

  // 5) bias + relu in place -> layer-2 input
  bias_relu_k<<<nblk((long long)N * 64, BS), BS, 0, stream>>>(agg1, b1, (long long)N * 64);

  // 6) layer-2 GEMM: h2 = relu(h) @ W2   [N,64]x[64,40], K=2x32, N=3x16 (padded 48)
  {
    int rtiles = (N + 15) / 16;
    gemm_wmma<2, 3><<<nblk(rtiles, 8), BS, 0, stream>>>(agg1, w2f, h2, N, 64, 40, 40);
  }

  // 7) attention dots, layer 2 (1 head x 40 ch)
  att2_k<<<nblk(N, BS), BS, 0, stream>>>(h2, as2w, ad2w, as2, ad2, N);

  // 8) edge softmax + aggregate, layer 2
  emax_k<<<nblk(Etot, BS), BS, 0, stream>>>(ei, E, Etot, as2, ad2, m2, 1);
  eden_k<<<nblk(Etot, BS), BS, 0, stream>>>(ei, E, Etot, as2, ad2, m2, den2, 1);
  eagg2_k<<<nblk((long long)Etot * 5, BS), BS, 0, stream>>>(ei, E, Etot, as2, ad2, m2, den2, h2, out2);

  // 9) bias + log_softmax -> d_out
  final_ls_k<<<nblk(N, BS), BS, 0, stream>>>(out2, b2, (float*)d_out, N);
}
